// LabelSmoothingCTCLoss_52956946759796
// MI455X (gfx1250) — compile-verified, matrix operand order checked by reading, not run on hardware
//
#include <hip/hip_runtime.h>

#define T_DIM 1024
#define B_DIM 128
#define C_DIM 256
#define L_DIM 64
#define S_DIM 129
#define NEGV  (-1e30f)
#define DEPTH 4

typedef __attribute__((ext_vector_type(16))) _Float16 v16h;
typedef __attribute__((ext_vector_type(8)))  float    v8f;

// ---------------------------------------------------------------- accumulators
__global__ void init_acc_kernel(float* acc) {
  acc[0] = 0.f;  // sum of per-batch CTC nll
  acc[1] = 0.f;  // sum over (t,b,c) of p*lp  (negative of total entropy)
}

// ---------------------------------------------------------------- entropy via WMMA
// Each wave handles 16 flat rows (t*B+b) x 256 channels.
// A-matrix (16x32 f16): lane l and lane l+16 both map to row l&15 with disjoint
// 16-channel halves; B = ones => D[m,n] = sum_k A[m,k] for every n.
// D layout: lane 0 holds rows 0..7 (c[0..7]), lane 16 holds rows 8..15.
__global__ __launch_bounds__(256) void entropy_wmma_kernel(const float* __restrict__ lp,
                                                           float* __restrict__ acc) {
  __shared__ float blk;
  if (threadIdx.x == 0) blk = 0.f;
  __syncthreads();

  const int lane = threadIdx.x & 31;
  const int wv   = threadIdx.x >> 5;                 // 0..7 waves per block
  const long tile = (long)blockIdx.x * 8 + wv;       // 16 rows per wave-tile
  const long row  = tile * 16 + (lane & 15);
  const float* rp = lp + row * C_DIM + ((lane >> 4) << 4);  // +16 cols for lanes>=16

  v16h ones;
  #pragma unroll
  for (int i = 0; i < 16; ++i) ones[i] = (_Float16)1.0f;

  v8f c = {};
  #pragma unroll
  for (int ch = 0; ch < 8; ++ch) {                   // 8 x 32 channels = 256
    const float4* p4 = (const float4*)(rp + ch * 32);
    float4 q0 = p4[0], q1 = p4[1], q2 = p4[2], q3 = p4[3];
    float x[16] = {q0.x,q0.y,q0.z,q0.w, q1.x,q1.y,q1.z,q1.w,
                   q2.x,q2.y,q2.z,q2.w, q3.x,q3.y,q3.z,q3.w};
    v16h a;
    #pragma unroll
    for (int i = 0; i < 16; ++i) a[i] = (_Float16)(__expf(x[i]) * x[i]);
    c = __builtin_amdgcn_wmma_f32_16x16x32_f16(false, a, false, ones,
                                               (short)0, c, false, false);
  }
  float s = c[0]+c[1]+c[2]+c[3]+c[4]+c[5]+c[6]+c[7];
  if ((lane & 15) == 0) atomicAdd(&blk, s);          // lanes 0 and 16: rows 0-7 / 8-15
  __syncthreads();
  if (threadIdx.x == 0) atomicAdd(&acc[1], blk);
}

// ---------------------------------------------------------------- async row staging
__device__ __forceinline__ void async_row_b128x2(unsigned lds_addr, unsigned goff,
                                                 const float* base) {
  // 32 lanes x 16B x 2 = 1024B row; offset applies to both LDS and global addr.
  asm volatile(
      "global_load_async_to_lds_b128 %0, %1, %2 offset:0\n\t"
      "global_load_async_to_lds_b128 %0, %1, %2 offset:512"
      :: "v"(lds_addr), "v"(goff), "s"(base) : "memory");
}
__device__ __forceinline__ void wait_async_le6() {
  asm volatile("s_wait_asynccnt 6" ::: "memory");    // 2*(DEPTH-1)
}

// ---------------------------------------------------------------- CTC forward DP
// One wave32 per batch; 4-5 states per lane; alpha in LDS (in-order per-wave DS).
__global__ __launch_bounds__(32) void ctc_kernel(const float* __restrict__ lp,
                                                 const int* __restrict__ targets,
                                                 const int* __restrict__ in_len,
                                                 const int* __restrict__ tg_len,
                                                 float* __restrict__ acc) {
  __shared__ __align__(16) float rowbuf[DEPTH][C_DIM];
  __shared__ float abuf[S_DIM + 3];                  // alpha[s] at abuf[s+2]; pads NEG

  const int b     = blockIdx.x;
  const int lane  = threadIdx.x;
  const int inlen = in_len[b];
  const int tl    = tg_len[b];
  const int* tg   = targets + b * L_DIM;

  const unsigned rowbase = (unsigned)(size_t)(void*)&rowbuf[0][0];
  const unsigned laneoff = (unsigned)lane * 16u;

  // prefetch rows 0..DEPTH-1
  #pragma unroll
  for (int r = 0; r < DEPTH; ++r)
    async_row_b128x2(rowbase + (unsigned)r * 1024u + laneoff,
                     (unsigned)((r * B_DIM + b) * (C_DIM * 4)) + laneoff, lp);

  // per-lane extended labels & skip-transition gates
  int   col[5]; bool al2[5]; float ak[5];
  #pragma unroll
  for (int k = 0; k < 5; ++k) {
    int s = lane + 32 * k;
    if (s < S_DIM) {
      if (s & 1) {
        int i = (s - 1) >> 1;
        col[k] = tg[i];
        al2[k] = (s == 1) ? true : (tg[i] != tg[i - 1]);  // s==1 reads NEG pad anyway
      } else { col[k] = 0; al2[k] = false; }
    }
  }
  if (lane < 2) abuf[lane] = NEGV;

  // t = 0 init
  wait_async_le6();                                  // row 0 resident
  const float* rp0 = &rowbuf[0][0];
  #pragma unroll
  for (int k = 0; k < 5; ++k) {
    int s = lane + 32 * k;
    if (s < S_DIM) {
      float v = (s < 2) ? rp0[col[k]] : NEGV;
      ak[k] = v;
      abuf[s + 2] = v;
    }
  }
  float ll = NEGV;
  if (inlen == 1 && lane == 0) {
    float a1 = abuf[2 * tl + 2], a0 = abuf[2 * tl + 1];
    float m = fmaxf(a1, a0);
    ll = m + __logf(__expf(a1 - m) + __expf(a0 - m));
  }

  for (int t = 1; t < inlen; ++t) {
    // refill slot (t-1)%DEPTH (consumed last iter) with row t+DEPTH-1 (clamped
    // so in-flight count stays uniform; clamped rows are never read).
    int tp = t + DEPTH - 1; if (tp > T_DIM - 1) tp = T_DIM - 1;
    async_row_b128x2(rowbase + (unsigned)((t - 1) & (DEPTH - 1)) * 1024u + laneoff,
                     (unsigned)((tp * B_DIM + b) * (C_DIM * 4)) + laneoff, lp);
    wait_async_le6();                                // row t resident
    const float* rp = &rowbuf[t & (DEPTH - 1)][0];

    // phase 1: all neighbor loads (whole-wave loads precede whole-wave stores)
    float p1v[5], p2v[5], lpv[5];
    #pragma unroll
    for (int k = 0; k < 5; ++k) {
      int s = lane + 32 * k;
      if (s < S_DIM) {
        p1v[k] = abuf[s + 1];
        p2v[k] = al2[k] ? abuf[s] : NEGV;
        lpv[k] = rp[col[k]];
      }
    }
    __builtin_amdgcn_wave_barrier();
    // phase 2: LSE update + store
    #pragma unroll
    for (int k = 0; k < 5; ++k) {
      int s = lane + 32 * k;
      if (s < S_DIM) {
        float a = ak[k], p1 = p1v[k], p2 = p2v[k];
        float m = fmaxf(a, fmaxf(p1, p2));
        float nv = m + __logf(__expf(a - m) + __expf(p1 - m) + __expf(p2 - m)) + lpv[k];
        ak[k] = nv;
        abuf[s + 2] = nv;
      }
    }
    __builtin_amdgcn_wave_barrier();
    if (t == inlen - 1 && lane == 0) {
      float a1 = abuf[2 * tl + 2], a0 = abuf[2 * tl + 1];
      float m = fmaxf(a1, a0);
      ll = m + __logf(__expf(a1 - m) + __expf(a0 - m));
    }
  }

  if (lane == 0) {
    float nll = -ll;
    if (nll > 1e29f) nll = 0.f;                      // zero_infinity
    atomicAdd(&acc[0], nll);
  }
}

// ---------------------------------------------------------------- combine
__global__ void finalize_kernel(const float* __restrict__ acc, float* __restrict__ out) {
  out[0] = 0.9f * (acc[0] / (float)B_DIM)
         + 0.1f * (acc[1] / (float)(T_DIM * B_DIM));  // acc[1] = -total entropy
}

// ---------------------------------------------------------------- launch
extern "C" void kernel_launch(void* const* d_in, const int* in_sizes, int n_in,
                              void* d_out, int out_size, void* d_ws, size_t ws_size,
                              hipStream_t stream) {
  const float* lp      = (const float*)d_in[0];
  const int*   targets = (const int*)d_in[1];
  const int*   in_len  = (const int*)d_in[2];
  const int*   tg_len  = (const int*)d_in[3];
  float* acc = (float*)d_ws;
  float* out = (float*)d_out;

  init_acc_kernel<<<1, 1, 0, stream>>>(acc);
  entropy_wmma_kernel<<<(T_DIM * B_DIM) / 128, 256, 0, stream>>>(lp, acc);
  ctc_kernel<<<B_DIM, 32, 0, stream>>>(lp, targets, in_len, tg_len, acc);
  finalize_kernel<<<1, 1, 0, stream>>>(acc, out);
}
